// MultiHeadLatentAttention_21887153340505
// MI455X (gfx1250) — matrix operand order repaired and checked by reading, not compiled, not run on hardware
//
#include <hip/hip_runtime.h>

// ---------------- problem constants ----------------
constexpr int Bc      = 2;
constexpr int Sc      = 2048;
constexpr int HIDc    = 2048;
constexpr int Hc      = 16;
constexpr int QLORA   = 1536;
constexpr int KVLORA  = 128;
constexpr int NOPE    = 128;
constexpr int ROPE    = 64;
constexpr int QKHEAD  = NOPE + ROPE;   // 192
constexpr int VHEAD   = 64;

typedef __attribute__((ext_vector_type(16))) __bf16 v16bf;
typedef __attribute__((ext_vector_type(8)))  float  v8f;

__device__ __forceinline__ unsigned short f2bf(float f) {
  __bf16 h = (__bf16)f;
  return __builtin_bit_cast(unsigned short, h);
}

// -------- WMMA fragment loaders (ISA 7.12.2, 16-bit layouts, wave32) --------
// A 16x32: lanes 0-15 row M=lane:   V0..3 -> K 0..7,  V4..7 -> K 16..23
//          lanes16-31 row M=lane-16: V0..3 -> K 8..15, V4..7 -> K 24..31
__device__ __forceinline__ v16bf load_frag_a(const unsigned short* base, int ld) {
  const int lane = threadIdx.x & 31;
  const unsigned short* row = base + (size_t)(lane & 15) * ld;
  const int khalf = (lane & 16) >> 1;             // 0 or 8
  union { v16bf v; unsigned int u[8]; } f;
#pragma unroll
  for (int j = 0; j < 8; ++j) {
    int k = ((j & 4) << 2) + khalf + ((j & 3) << 1); // two 16B runs @ shorts 0.. and 16..
    f.u[j] = *reinterpret_cast<const unsigned int*>(row + k);
  }
  return f.v;
}
// B 32x16: lanes 0-15 col N=lane: V0..3 -> K 0..7, V4..7 -> K 16..23; lanes 16-31: K 8..15, 24..31.
// Bs stored [N][K] so this is two 16B runs per lane.
__device__ __forceinline__ v16bf load_frag_b(const unsigned short* base, int ld) {
  const int lane = threadIdx.x & 31;
  const unsigned short* row = base + (size_t)(lane & 15) * ld;
  const int kbase = (lane & 16) >> 1;
  union { v16bf v; unsigned int u[8]; } f;
#pragma unroll
  for (int j = 0; j < 8; ++j)
    f.u[j] = *reinterpret_cast<const unsigned int*>(row + kbase + ((j & 4) << 2) + ((j & 3) << 1));
  return f.v;
}

__device__ __forceinline__ void store_tile(float* C, int ldc, int row0, int col,
                                           const v8f& acc, float alpha) {
#pragma unroll
  for (int r = 0; r < 8; ++r)
    C[(size_t)(row0 + r) * ldc + col] = alpha * acc[r];
}

// ---------------- generic batched WMMA GEMM, double-buffered ----------------
// C[M,N](f32) = alpha * A[M,K] * B[K,N]    (TRANS_B: B is row-major [N,K])
// NT=128: 8 waves 2x4, 64x32/wave (8 wmma : 6 frag loads). NT=64: 8 waves 4x2, 32x32/wave.
// Requires M%128==0, N%NT==0, K%32==0.
template <int NT, bool TRANS_B, bool A_IS_F32>
__global__ void __launch_bounds__(256)
gemm_bf16_wmma(const void* __restrict__ Ap, const unsigned short* __restrict__ Bp,
               float* __restrict__ Cp, int M, int N, int K,
               int lda, int ldb, int ldc,
               long long sA, long long sB, long long sC, float alpha) {
  constexpr int MT = 128, KT = 32, LDK = 40;            // LDK pad kills bank conflicts
  constexpr int TM = (NT == 128) ? 4 : 2;               // 16-row frags per wave
  constexpr int TN = 2;                                 // 16-col frags per wave
  __shared__ __align__(16) unsigned short As[2][MT * LDK];
  __shared__ __align__(16) unsigned short Bs[2][NT * LDK];

  const int tid  = threadIdx.x;
  const int lane = tid & 31;
  const int wave = tid >> 5;
  const int wm = (NT == 128) ? (wave >> 2) * 64 : (wave >> 1) * 32;
  const int wn = (NT == 128) ? (wave & 3) * 32 : (wave & 1) * 32;
  const int m0 = blockIdx.y * MT;
  const int n0 = blockIdx.x * NT;
  const int bz = blockIdx.z;

  const float*          Af = (const float*)Ap + sA * bz;
  const unsigned short* Ab = (const unsigned short*)Ap + sA * bz;
  const unsigned short* Bg = Bp + sB * bz;
  float*                Cg = Cp + sC * bz;

  // staging assignments (one uint4 per thread for A; B: 1 (trans) or 2 (k-pair) uint4)
  const int arow = tid >> 1, aseg = (tid & 1) * 8;
  const int brow = (NT == 128) ? (tid >> 1) : ((tid & 127) >> 1);   // TRANS_B
  const int bsg  = (tid & 1) * 8;
  const int bk   = (NT == 128) ? ((tid >> 4) * 2) : (((tid & 127) >> 3) * 2);
  const int bn   = (NT == 128) ? ((tid & 15) * 8) : ((tid & 7) * 8);
  const bool bact = (NT == 128) || (tid < 128);

  uint4 ra{}; float4 raf0{}, raf1{}; uint4 rb0{}, rb1{};

  auto gload = [&](int k0) {
    if constexpr (A_IS_F32) {
      const float* src = Af + (size_t)(m0 + arow) * lda + k0 + aseg;
      raf0 = *(const float4*)src; raf1 = *(const float4*)(src + 4);
    } else {
      ra = *(const uint4*)(Ab + (size_t)(m0 + arow) * lda + k0 + aseg);
    }
    if constexpr (TRANS_B) {
      if (bact) rb0 = *(const uint4*)(Bg + (size_t)(n0 + brow) * ldb + k0 + bsg);
    } else {
      if (bact) {
        rb0 = *(const uint4*)(Bg + (size_t)(k0 + bk) * ldb + n0 + bn);
        rb1 = *(const uint4*)(Bg + (size_t)(k0 + bk + 1) * ldb + n0 + bn);
      }
    }
  };
  auto stage = [&](int buf) {
    if constexpr (A_IS_F32) {
      unsigned short* d = &As[buf][arow * LDK + aseg];
      d[0] = f2bf(raf0.x); d[1] = f2bf(raf0.y); d[2] = f2bf(raf0.z); d[3] = f2bf(raf0.w);
      d[4] = f2bf(raf1.x); d[5] = f2bf(raf1.y); d[6] = f2bf(raf1.z); d[7] = f2bf(raf1.w);
    } else {
      *(uint4*)&As[buf][arow * LDK + aseg] = ra;
    }
    if constexpr (TRANS_B) {
      if (bact) *(uint4*)&Bs[buf][brow * LDK + bsg] = rb0;
    } else {
      if (bact) {
        unsigned short lo[8], hi[8];
        *(uint4*)lo = rb0; *(uint4*)hi = rb1;
#pragma unroll
        for (int i = 0; i < 8; ++i) {
          unsigned int pk = (unsigned int)lo[i] | ((unsigned int)hi[i] << 16);
          *(unsigned int*)&Bs[buf][(bn + i) * LDK + bk] = pk;   // K-pair packed b32
        }
      }
    }
  };

  v8f acc[TM][TN] = {};

  gload(0);
  stage(0);
  __syncthreads();

  const int nk = K / KT;
  for (int ki = 0; ki < nk; ++ki) {
    const int cur = ki & 1;
    if (ki + 1 < nk) gload((ki + 1) * KT);    // prefetch next tile into regs

    v16bf af[TM], bfr[TN];
#pragma unroll
    for (int i = 0; i < TM; ++i) af[i] = load_frag_a(&As[cur][(wm + i * 16) * LDK], LDK);
#pragma unroll
    for (int j = 0; j < TN; ++j) bfr[j] = load_frag_b(&Bs[cur][(wn + j * 16) * LDK], LDK);
#pragma unroll
    for (int i = 0; i < TM; ++i)
#pragma unroll
      for (int j = 0; j < TN; ++j)
        acc[i][j] = __builtin_amdgcn_wmma_f32_16x16x32_bf16(
            false, af[i], false, bfr[j], (short)0, acc[i][j], false, false);

    if (ki + 1 < nk) {
      stage(cur ^ 1);
      __syncthreads();
    }
  }

  const int col0 = n0 + wn + (lane & 15);
  const int rb_  = m0 + wm + ((lane >> 4) << 3);  // C/D: VGPR r -> M = r + 8*(lane>=16)
#pragma unroll
  for (int i = 0; i < TM; ++i)
#pragma unroll
    for (int j = 0; j < TN; ++j)
      store_tile(Cg, ldc, rb_ + i * 16, col0 + j * 16, acc[i][j], alpha);
}

// ---------------- elementwise / reduction kernels ----------------
__global__ void __launch_bounds__(256)
f32_to_bf16_kernel(const float* __restrict__ x, unsigned short* __restrict__ y, long long n) {
  long long i = (long long)blockIdx.x * blockDim.x + threadIdx.x;
  long long stride = (long long)gridDim.x * blockDim.x;
  for (; i < n; i += stride) y[i] = f2bf(x[i]);
}

__global__ void __launch_bounds__(256)
rmsnorm_bf16_kernel(const float* __restrict__ x, int ldx, const float* __restrict__ w,
                    unsigned short* __restrict__ y, int ldy, int N) {
  __shared__ float red[256];
  const int row = blockIdx.x;
  const float* xr = x + (size_t)row * ldx;
  float s = 0.f;
  for (int i = threadIdx.x; i < N; i += 256) { float v = xr[i]; s += v * v; }
  red[threadIdx.x] = s; __syncthreads();
  for (int off = 128; off > 0; off >>= 1) {
    if (threadIdx.x < off) red[threadIdx.x] += red[threadIdx.x + off];
    __syncthreads();
  }
  const float inv = rsqrtf(red[0] / (float)N + 1e-6f);
  for (int i = threadIdx.x; i < N; i += 256)
    y[(size_t)row * ldy + i] = f2bf(xr[i] * inv * w[i]);
}

// in-place row softmax, n % 1024 == 0, float4 vectorized
__global__ void __launch_bounds__(256)
softmax_rows_kernel(float* __restrict__ p, int n) {
  __shared__ float red[256];
  float4* r = (float4*)(p + (size_t)blockIdx.x * n);
  const int nv = n >> 2;
  float m = -3.4e38f;
  for (int i = threadIdx.x; i < nv; i += 256) {
    float4 v = r[i];
    m = fmaxf(m, fmaxf(fmaxf(v.x, v.y), fmaxf(v.z, v.w)));
  }
  red[threadIdx.x] = m; __syncthreads();
  for (int off = 128; off > 0; off >>= 1) {
    if (threadIdx.x < off) red[threadIdx.x] = fmaxf(red[threadIdx.x], red[threadIdx.x + off]);
    __syncthreads();
  }
  m = red[0]; __syncthreads();
  float s = 0.f;
  for (int i = threadIdx.x; i < nv; i += 256) {
    float4 v = r[i];
    v.x = __expf(v.x - m); v.y = __expf(v.y - m);
    v.z = __expf(v.z - m); v.w = __expf(v.w - m);
    s += (v.x + v.y) + (v.z + v.w);
    r[i] = v;
  }
  red[threadIdx.x] = s; __syncthreads();
  for (int off = 128; off > 0; off >>= 1) {
    if (threadIdx.x < off) red[threadIdx.x] += red[threadIdx.x + off];
    __syncthreads();
  }
  const float inv = 1.f / red[0];
  for (int i = threadIdx.x; i < nv; i += 256) {
    float4 v = r[i];
    v.x *= inv; v.y *= inv; v.z *= inv; v.w *= inv;
    r[i] = v;
  }
}

// q (B,S,H,192) f32 + RoPE -> query (B,H,S,192) bf16.  grid(S,H,B), block 192
__global__ void __launch_bounds__(192)
build_query_kernel(const float* __restrict__ q, const float* __restrict__ cosb,
                   const float* __restrict__ sinb, unsigned short* __restrict__ query) {
  const int s = blockIdx.x, h = blockIdx.y, b = blockIdx.z, d = threadIdx.x;
  const float* qp = q + (((size_t)b * Sc + s) * Hc + h) * QKHEAD;
  float val;
  if (d < NOPE) {
    val = qp[d];
  } else {
    const int r = d - NOPE;
    const size_t cs = ((size_t)b * Sc + s) * ROPE + r;
    const float x = qp[NOPE + r];
    const float other = (r < ROPE / 2) ? -qp[NOPE + r + ROPE / 2] : qp[NOPE + r - ROPE / 2];
    val = x * cosb[cs] + other * sinb[cs];
  }
  query[(((size_t)b * Hc + h) * Sc + s) * QKHEAD + d] = f2bf(val);
}

__global__ void __launch_bounds__(192)
build_keys_v_kernel(const float* __restrict__ kv, const float* __restrict__ ckv,
                    const float* __restrict__ cosb, const float* __restrict__ sinb,
                    unsigned short* __restrict__ keys, unsigned short* __restrict__ vout) {
  const int s = blockIdx.x, h = blockIdx.y, b = blockIdx.z, d = threadIdx.x;
  const float* kvp = kv + (((size_t)b * Sc + s) * Hc + h) * (NOPE + VHEAD);
  float val;
  if (d < NOPE) {
    val = kvp[d];
  } else {
    const int r = d - NOPE;
    const float* kr = ckv + ((size_t)b * Sc + s) * (KVLORA + ROPE) + KVLORA;
    const size_t cs = ((size_t)b * Sc + s) * ROPE + r;
    const float x = kr[r];
    const float other = (r < ROPE / 2) ? -kr[r + ROPE / 2] : kr[r - ROPE / 2];
    val = x * cosb[cs] + other * sinb[cs];
  }
  keys[(((size_t)b * Hc + h) * Sc + s) * QKHEAD + d] = f2bf(val);
  if (d < VHEAD)
    vout[(((size_t)b * Hc + h) * Sc + s) * VHEAD + d] = f2bf(kvp[NOPE + d]);
}

__global__ void __launch_bounds__(64)
attn_reorder_kernel(const float* __restrict__ a, unsigned short* __restrict__ y) {
  const int s = blockIdx.x, h = blockIdx.y, b = blockIdx.z, d = threadIdx.x;
  y[(((size_t)b * Sc + s) * Hc + h) * VHEAD + d] =
      f2bf(a[(((size_t)b * Hc + h) * Sc + s) * VHEAD + d]);
}

// ---------------- host orchestration ----------------
extern "C" void kernel_launch(void* const* d_in, const int* in_sizes, int n_in,
                              void* d_out, int out_size, void* d_ws, size_t ws_size,
                              hipStream_t stream) {
  const float* hidden  = (const float*)d_in[0];
  const float* cosb    = (const float*)d_in[1];
  const float* sinb    = (const float*)d_in[2];
  const float* q_a_w   = (const float*)d_in[3];
  const float* q_a_ln  = (const float*)d_in[4];
  const float* q_b_w   = (const float*)d_in[5];
  const float* kv_a_w  = (const float*)d_in[6];
  const float* kv_a_ln = (const float*)d_in[7];
  const float* kv_b_w  = (const float*)d_in[8];
  const float* o_w     = (const float*)d_in[9];

  const int BS = Bc * Sc;
  float* outp  = (float*)d_out;
  float* attnP = outp + (size_t)BS * HIDc;   // (B,H,S,S) probs live in d_out

  char* ws = (char*)d_ws;
  size_t off = 0;
  auto alloc = [&](size_t bytes) -> void* {
    void* p = ws + off;
    off = (off + bytes + 255) & ~(size_t)255;
    return p;
  };
  unsigned short* hid_bf  = (unsigned short*)alloc((size_t)BS * HIDc * 2);
  unsigned short* qaw_bf  = (unsigned short*)alloc((size_t)HIDc * QLORA * 2);
  unsigned short* qbw_bf  = (unsigned short*)alloc((size_t)QLORA * Hc * QKHEAD * 2);
  unsigned short* kvaw_bf = (unsigned short*)alloc((size_t)HIDc * (KVLORA + ROPE) * 2);
  unsigned short* kvbw_bf = (unsigned short*)alloc((size_t)KVLORA * Hc * (NOPE + VHEAD) * 2);
  unsigned short* ow_bf   = (unsigned short*)alloc((size_t)Hc * VHEAD * HIDc * 2);
  float*          qa_f    = (float*)alloc((size_t)BS * QLORA * 4);
  float*          ckv_f   = (float*)alloc((size_t)BS * (KVLORA + ROPE) * 4);
  unsigned short* qan_bf  = (unsigned short*)alloc((size_t)BS * QLORA * 2);
  unsigned short* ckvn_bf = (unsigned short*)alloc((size_t)BS * KVLORA * 2);
  float*          q_f     = (float*)alloc((size_t)BS * Hc * QKHEAD * 4);
  float*          kv_f    = (float*)alloc((size_t)BS * Hc * (NOPE + VHEAD) * 4);
  unsigned short* query_bf = (unsigned short*)alloc((size_t)Bc * Hc * Sc * QKHEAD * 2);
  unsigned short* keys_bf  = (unsigned short*)alloc((size_t)Bc * Hc * Sc * QKHEAD * 2);
  unsigned short* v_bf     = (unsigned short*)alloc((size_t)Bc * Hc * Sc * VHEAD * 2);
  float*          attno_f  = (float*)alloc((size_t)Bc * Hc * Sc * VHEAD * 4);
  unsigned short* attnr_bf = (unsigned short*)alloc((size_t)BS * Hc * VHEAD * 2);

  auto cvt = [&](const float* x, unsigned short* y, long long n) {
    f32_to_bf16_kernel<<<1024, 256, 0, stream>>>(x, y, n);
  };
  cvt(hidden, hid_bf,  (long long)BS * HIDc);
  cvt(q_a_w,  qaw_bf,  (long long)HIDc * QLORA);
  cvt(q_b_w,  qbw_bf,  (long long)QLORA * Hc * QKHEAD);
  cvt(kv_a_w, kvaw_bf, (long long)HIDc * (KVLORA + ROPE));
  cvt(kv_b_w, kvbw_bf, (long long)KVLORA * Hc * (NOPE + VHEAD));
  cvt(o_w,    ow_bf,   (long long)Hc * VHEAD * HIDc);

  // q_a = hidden @ q_a_w            [4096 x 1536, K=2048]
  gemm_bf16_wmma<128, false, false><<<dim3(QLORA / 128, BS / 128, 1), 256, 0, stream>>>(
      hid_bf, qaw_bf, qa_f, BS, QLORA, HIDc, HIDc, QLORA, QLORA, 0, 0, 0, 1.0f);
  // ckv = hidden @ kv_a_w           [4096 x 192, K=2048]  (small-N path)
  gemm_bf16_wmma<64, false, false><<<dim3((KVLORA + ROPE) / 64, BS / 128, 1), 256, 0, stream>>>(
      hid_bf, kvaw_bf, ckv_f, BS, KVLORA + ROPE, HIDc, HIDc, KVLORA + ROPE, KVLORA + ROPE,
      0, 0, 0, 1.0f);
  rmsnorm_bf16_kernel<<<BS, 256, 0, stream>>>(qa_f, QLORA, q_a_ln, qan_bf, QLORA, QLORA);
  rmsnorm_bf16_kernel<<<BS, 256, 0, stream>>>(ckv_f, KVLORA + ROPE, kv_a_ln, ckvn_bf, KVLORA, KVLORA);
  // q = qan @ q_b_w                 [4096 x 3072, K=1536]
  gemm_bf16_wmma<128, false, false><<<dim3(Hc * QKHEAD / 128, BS / 128, 1), 256, 0, stream>>>(
      qan_bf, qbw_bf, q_f, BS, Hc * QKHEAD, QLORA, QLORA, Hc * QKHEAD, Hc * QKHEAD,
      0, 0, 0, 1.0f);
  // kv = ckvn @ kv_b_w              [4096 x 3072, K=128]
  gemm_bf16_wmma<128, false, false><<<dim3(Hc * (NOPE + VHEAD) / 128, BS / 128, 1), 256, 0, stream>>>(
      ckvn_bf, kvbw_bf, kv_f, BS, Hc * (NOPE + VHEAD), KVLORA, KVLORA,
      Hc * (NOPE + VHEAD), Hc * (NOPE + VHEAD), 0, 0, 0, 1.0f);
  build_query_kernel<<<dim3(Sc, Hc, Bc), 192, 0, stream>>>(q_f, cosb, sinb, query_bf);
  build_keys_v_kernel<<<dim3(Sc, Hc, Bc), 192, 0, stream>>>(kv_f, ckv_f, cosb, sinb, keys_bf, v_bf);
  // scores = scaling * Q @ K^T -> d_out attn region  [per (b,h): 2048x2048, K=192]
  const float scaling = 1.0f / sqrtf((float)QKHEAD);
  gemm_bf16_wmma<128, true, false><<<dim3(Sc / 128, Sc / 128, Bc * Hc), 256, 0, stream>>>(
      query_bf, keys_bf, attnP, Sc, Sc, QKHEAD, QKHEAD, QKHEAD, Sc,
      (long long)Sc * QKHEAD, (long long)Sc * QKHEAD, (long long)Sc * Sc, scaling);
  softmax_rows_kernel<<<Bc * Hc * Sc, 256, 0, stream>>>(attnP, Sc);
  // attn = P @ V (A f32 converted in staging)  [2048x64, K=2048]  (small-N path)
  gemm_bf16_wmma<64, false, true><<<dim3(VHEAD / 64, Sc / 128, Bc * Hc), 256, 0, stream>>>(
      attnP, v_bf, attno_f, Sc, VHEAD, Sc, Sc, VHEAD, VHEAD,
      (long long)Sc * Sc, (long long)Sc * VHEAD, (long long)Sc * VHEAD, 1.0f);
  attn_reorder_kernel<<<dim3(Sc, Hc, Bc), 64, 0, stream>>>(attno_f, attnr_bf);
  // out = attn @ o_w                [4096 x 2048, K=1024]
  gemm_bf16_wmma<128, false, false><<<dim3(HIDc / 128, BS / 128, 1), 256, 0, stream>>>(
      attnr_bf, ow_bf, outp, BS, HIDc, Hc * VHEAD, Hc * VHEAD, HIDc, HIDc, 0, 0, 0, 1.0f);

  (void)in_sizes; (void)n_in; (void)out_size; (void)ws_size;
}
